// RefinementNetworkWithContext_32057635898162
// MI455X (gfx1250) — matrix operand order repaired and checked by reading, not compiled, not run on hardware
//
#include <hip/hip_runtime.h>
#include <hip/hip_bf16.h>
#include <math.h>

// ---------------------------------------------------------------------------
// RefinementNetworkWithContext on gfx1250 (MI455X)
//   Phase 1: brute-force kNN (VALU, register top-16 insertion sort)
//   Phase 2: per-point MLP via v_wmma_f32_16x16x32_f16 (f16 in, f32 accum)
// ---------------------------------------------------------------------------

#define N_PTS   16384
#define KNN_K   16
#define UPF     6

typedef __attribute__((ext_vector_type(16))) _Float16 v16h;
typedef __attribute__((ext_vector_type(8)))  float    v8f;

union Frag { v16h v; unsigned u[8]; };

// ---------------------------- weight conversion ----------------------------
// Stores each weight matrix transposed + zero-padded as f16: wT[j*CinPad + i]
// Offsets (in halfs): w1T 0, w2T 4096, w3T 12288, w4T 45056, w5T 110592,
// whT 143360; total 147456 halfs = 294912 bytes.
__global__ void cvt_weights_kernel(const float* __restrict__ w1,
                                   const float* __restrict__ w2,
                                   const float* __restrict__ w3,
                                   const float* __restrict__ w4,
                                   const float* __restrict__ w5,
                                   const float* __restrict__ wh,
                                   _Float16* __restrict__ wbuf) {
    const int Cin[6]   = {51, 64, 128, 256, 256, 128};
    const int Cout[6]  = {64, 128, 256, 256, 128, 18};
    const int CinP[6]  = {64, 64, 128, 256, 256, 128};
    const int CoutP[6] = {64, 128, 256, 256, 128, 32};
    const int off[6]   = {0, 4096, 12288, 45056, 110592, 143360};
    const float* src[6] = {w1, w2, w3, w4, w5, wh};

    int m  = blockIdx.y;
    int id = blockIdx.x * blockDim.x + threadIdx.x;
    int n  = CinP[m] * CoutP[m];
    if (id >= n) return;
    int j = id / CinP[m];          // output column of original W
    int i = id - j * CinP[m];      // input row of original W
    float v = (i < Cin[m] && j < Cout[m]) ? src[m][i * Cout[m] + j] : 0.0f;
    wbuf[off[m] + id] = (_Float16)v;
}

// --------------------------------- kNN -------------------------------------
// One thread per query point; candidates staged through LDS in chunks of 256.
// Keeps a sorted (ascending d2) top-16 in registers; ties keep lower index
// first (matches jax.lax.top_k tie-breaking since we scan in index order).
__global__ void knn_kernel(const float* __restrict__ pts,
                           int* __restrict__ idx) {
    __shared__ float4 cand[256];
    const int qi = blockIdx.x * 256 + threadIdx.x;
    const float qx = pts[qi * 3 + 0];
    const float qy = pts[qi * 3 + 1];
    const float qz = pts[qi * 3 + 2];
    const float qs = qx * qx + qy * qy + qz * qz;

    float bd[KNN_K];
    int   bi[KNN_K];
#pragma unroll
    for (int t = 0; t < KNN_K; ++t) { bd[t] = 3.4e38f; bi[t] = 0; }

    for (int j0 = 0; j0 < N_PTS; j0 += 256) {
        __syncthreads();
        {
            int j = j0 + threadIdx.x;
            float cx = pts[j * 3 + 0];
            float cy = pts[j * 3 + 1];
            float cz = pts[j * 3 + 2];
            cand[threadIdx.x] = make_float4(cx, cy, cz,
                                            cx * cx + cy * cy + cz * cz);
        }
        __syncthreads();
        for (int jj = 0; jj < 256; ++jj) {
            float4 c = cand[jj];
            // same algebraic form as the reference: |q|^2 + |c|^2 - 2 q.c
            float d2 = qs + c.w - 2.0f * (qx * c.x + qy * c.y + qz * c.z);
            int j2 = j0 + jj;
            if (j2 != qi && d2 < bd[KNN_K - 1]) {
#pragma unroll
                for (int t = KNN_K - 1; t >= 0; --t) {
                    if (bd[t] > d2) {
                        if (t > 0 && bd[t - 1] > d2) {
                            bd[t] = bd[t - 1]; bi[t] = bi[t - 1];
                        } else {
                            bd[t] = d2; bi[t] = j2;
                        }
                    }
                }
            }
        }
    }
#pragma unroll
    for (int t = 0; t < KNN_K; ++t) idx[qi * KNN_K + t] = bi[t];
}

// --------------------------------- MLP -------------------------------------
#define XPITCH 264   // f16 activation row pitch (covers max Cin=256, even)
#define YPITCH 260   // f32 pre-activation row pitch
#define WAVES  2     // waves per block (LDS: ~49 KB)

struct MlpParams {
    const float* pts;
    const int*   idx;
    const _Float16* wbuf;
    const float *b1, *g1, *be1;
    const float *b2, *g2, *be2;
    const float *b3, *g3, *be3;
    const float *b4, *g4, *be4;
    const float *b5, *g5, *be5;
    const float *bh;
    float* out;
};

// One layer: y = x @ W + b (WMMA), then LayerNorm + exact GELU back into xh.
__device__ __forceinline__ void mlp_layer(const _Float16* __restrict__ wT,
                                          const float* __restrict__ bias,
                                          const float* __restrict__ gam,
                                          const float* __restrict__ bet,
                                          int CinPad, int Cout,
                                          _Float16* xh, float* yv, int lane) {
    const int m16 = lane & 15;
    const int hi  = lane >> 4;          // 0 or 1 (lane half)

    for (int n0 = 0; n0 < Cout; n0 += 16) {
        v8f c = {};
        for (int k0 = 0; k0 < CinPad; k0 += 32) {
            Frag a, b;
#pragma unroll
            for (int r = 0; r < 8; ++r) {
                // A: this code's contiguous k order, K = 16*hi + 2*r (B matches)
                int ka = k0 + (hi << 4) + (r << 1);
                a.u[r] = *(const unsigned*)(xh + m16 * XPITCH + ka);
                // B: same contiguous k order as A, K = 16*hi + 2*r, N = lane&15
                int kb = k0 + (hi << 4) + (r << 1);
                b.u[r] = *(const unsigned*)(wT + (n0 + m16) * CinPad + kb);
            }
            c = __builtin_amdgcn_wmma_f32_16x16x32_f16(
                    false, a.v, false, b.v, (short)0, c, false, false);
        }
        float bv = bias[n0 + m16];
#pragma unroll
        for (int r = 0; r < 8; ++r)
            yv[(r + (hi << 3)) * YPITCH + n0 + m16] = c[r] + bv;
    }
    __syncthreads();

    // LayerNorm (ddof=0) + gelu(approximate=False); lanes 0..15: one row each
    if (lane < 16) {
        const float inv = 1.0f / (float)Cout;
        float s = 0.0f;
        for (int cc = 0; cc < Cout; ++cc) s += yv[lane * YPITCH + cc];
        float mu = s * inv;
        float vv = 0.0f;
        for (int cc = 0; cc < Cout; ++cc) {
            float d = yv[lane * YPITCH + cc] - mu;
            vv += d * d;
        }
        float rstd = rsqrtf(vv * inv + 1e-5f);
        for (int cc = 0; cc < Cout; ++cc) {
            float t = (yv[lane * YPITCH + cc] - mu) * rstd * gam[cc] + bet[cc];
            float g = 0.5f * t * (1.0f + erff(t * 0.70710678118654752f));
            xh[lane * XPITCH + cc] = (_Float16)g;
        }
    }
    __syncthreads();
}

__global__ void mlp_kernel(MlpParams P) {
    __shared__ _Float16 sxh[WAVES][16 * XPITCH];
    __shared__ float    syv[WAVES][16 * YPITCH];

    const int lane = threadIdx.x & 31;
    const int wave = threadIdx.x >> 5;
    const int tile = blockIdx.x * WAVES + wave;
    const int p0   = tile * 16;

    _Float16* xh = sxh[wave];
    float*    yv = syv[wave];

    // ---- build context rows: [p(3), deltas(48)], zero-pad to 64 ----
    if (lane < 16) {
        int p = p0 + lane;
        float px = P.pts[p * 3 + 0];
        float py = P.pts[p * 3 + 1];
        float pz = P.pts[p * 3 + 2];
        _Float16* row = xh + lane * XPITCH;
        row[0] = (_Float16)px; row[1] = (_Float16)py; row[2] = (_Float16)pz;
        for (int k = 0; k < KNN_K; ++k) {
            int j = P.idx[p * KNN_K + k];
            row[3 + 3 * k + 0] = (_Float16)(P.pts[j * 3 + 0] - px);
            row[3 + 3 * k + 1] = (_Float16)(P.pts[j * 3 + 1] - py);
            row[3 + 3 * k + 2] = (_Float16)(P.pts[j * 3 + 2] - pz);
        }
        for (int cc = 51; cc < 64; ++cc) row[cc] = (_Float16)0.0f;
    }
    __syncthreads();

    const _Float16* wbuf = P.wbuf;
    mlp_layer(wbuf + 0,      P.b1, P.g1, P.be1,  64,  64, xh, yv, lane);
    mlp_layer(wbuf + 4096,   P.b2, P.g2, P.be2,  64, 128, xh, yv, lane);
    mlp_layer(wbuf + 12288,  P.b3, P.g3, P.be3, 128, 256, xh, yv, lane);
    mlp_layer(wbuf + 45056,  P.b4, P.g4, P.be4, 256, 256, xh, yv, lane);
    mlp_layer(wbuf + 110592, P.b5, P.g5, P.be5, 256, 128, xh, yv, lane);

    // ---- head: 128 -> 18 (padded to 32), write refined points ----
    const _Float16* whT = wbuf + 143360;
    const int m16 = lane & 15;
    const int hi  = lane >> 4;
    for (int n0 = 0; n0 < 32; n0 += 16) {
        v8f c = {};
        for (int k0 = 0; k0 < 128; k0 += 32) {
            Frag a, b;
#pragma unroll
            for (int r = 0; r < 8; ++r) {
                int ka = k0 + (hi << 4) + (r << 1);
                a.u[r] = *(const unsigned*)(xh + m16 * XPITCH + ka);
                int kb = k0 + (hi << 4) + (r << 1);
                b.u[r] = *(const unsigned*)(whT + (n0 + m16) * 128 + kb);
            }
            c = __builtin_amdgcn_wmma_f32_16x16x32_f16(
                    false, a.v, false, b.v, (short)0, c, false, false);
        }
        int col = n0 + m16;
        if (col < UPF * 3) {
            float bb = P.bh[col];
            int u  = col / 3;
            int cc = col - u * 3;
#pragma unroll
            for (int r = 0; r < 8; ++r) {
                int p = p0 + r + (hi << 3);
                P.out[(p * UPF + u) * 3 + cc] = P.pts[p * 3 + cc] + c[r] + bb;
            }
        }
    }
}

// ------------------------------- launcher ----------------------------------
extern "C" void kernel_launch(void* const* d_in, const int* in_sizes, int n_in,
                              void* d_out, int out_size, void* d_ws, size_t ws_size,
                              hipStream_t stream) {
    (void)in_sizes; (void)n_in; (void)out_size; (void)ws_size;

    const float* pts = (const float*)d_in[0];
    const float* w1 = (const float*)d_in[1];
    const float* b1 = (const float*)d_in[2];
    const float* g1 = (const float*)d_in[3];
    const float* be1 = (const float*)d_in[4];
    const float* w2 = (const float*)d_in[5];
    const float* b2 = (const float*)d_in[6];
    const float* g2 = (const float*)d_in[7];
    const float* be2 = (const float*)d_in[8];
    const float* w3 = (const float*)d_in[9];
    const float* b3 = (const float*)d_in[10];
    const float* g3 = (const float*)d_in[11];
    const float* be3 = (const float*)d_in[12];
    const float* w4 = (const float*)d_in[13];
    const float* b4 = (const float*)d_in[14];
    const float* g4 = (const float*)d_in[15];
    const float* be4 = (const float*)d_in[16];
    const float* w5 = (const float*)d_in[17];
    const float* b5 = (const float*)d_in[18];
    const float* g5 = (const float*)d_in[19];
    const float* be5 = (const float*)d_in[20];
    const float* wh = (const float*)d_in[21];
    const float* bh = (const float*)d_in[22];

    _Float16* wbuf = (_Float16*)d_ws;                 // 294912 bytes of f16 weights
    int* idx = (int*)((char*)d_ws + 294912);          // N*K ints = 1 MB

    // 1) convert+pad+transpose weights to f16 (max matrix 65536 elems)
    cvt_weights_kernel<<<dim3(256, 6), 256, 0, stream>>>(w1, w2, w3, w4, w5, wh, wbuf);

    // 2) brute-force kNN
    knn_kernel<<<N_PTS / 256, 256, 0, stream>>>(pts, idx);

    // 3) MLP + head via WMMA; one wave per 16-point tile
    MlpParams P;
    P.pts = pts; P.idx = idx; P.wbuf = wbuf;
    P.b1 = b1; P.g1 = g1; P.be1 = be1;
    P.b2 = b2; P.g2 = g2; P.be2 = be2;
    P.b3 = b3; P.g3 = g3; P.be3 = be3;
    P.b4 = b4; P.g4 = g4; P.be4 = be4;
    P.b5 = b5; P.g5 = g5; P.be5 = be5;
    P.bh = bh; P.out = (float*)d_out;
    mlp_kernel<<<(N_PTS / 16) / WAVES, WAVES * 32, 0, stream>>>(P);
}
